// DecoderAttention3_64450279244329
// MI455X (gfx1250) — compile-verified, matrix operand order checked by reading, not run on hardware
//
#include <hip/hip_runtime.h>

// ---------------------------------------------------------------------------
// DecoderAttention3 for MI455X (gfx1250): bf16 WMMA GEMMs + fused LSTM step
// B=64 T=60 S=60 H=512 V=32000 L=2
// Round 2: 128x128 block tile + GLOBAL_LOAD_ASYNC_TO_LDS_B128 staging.
// ---------------------------------------------------------------------------

typedef __attribute__((ext_vector_type(16))) __bf16 v16bf;
typedef __attribute__((ext_vector_type(8)))  float  v8f;

static constexpr int Bb = 64, Tt = 60, Ss = 60, Hh = 512, Vv = 32000, Ll = 2;
static constexpr float NEGMASK = -1.0e9f;

#define WMMA_BF16(a, b, c) \
  __builtin_amdgcn_wmma_f32_16x16x32_bf16(false, (a), false, (b), (short)0, (c), false, false)

__device__ __forceinline__ float sigmoidf_(float x) { return 1.0f / (1.0f + __expf(-x)); }

// Async copy of 16B from global to LDS (per-lane), tracked by ASYNCcnt.
__device__ __forceinline__ void async_copy_b128(unsigned lds_off, const void* gsrc) {
  unsigned long long ga = (unsigned long long)(uintptr_t)gsrc;
  asm volatile("global_load_async_to_lds_b128 %0, %1, off"
               :: "v"(lds_off), "v"(ga) : "memory");
}
__device__ __forceinline__ void wait_async0() {
#if __has_builtin(__builtin_amdgcn_s_wait_asynccnt)
  __builtin_amdgcn_s_wait_asynccnt(0);
#else
  asm volatile("s_wait_asynccnt 0x0" ::: "memory");
#endif
}

// Load a 16x32 bf16 fragment (A-matrix layout, ISA 7.12.2). Works for both
// A ([M,K] row-major) and B when the weight is stored [N,K] row-major
// (i.e. B in column-major K x N): same lane/K packing.
//   lane<16 : row=lane,    K = {0..7, 16..23}
//   lane>=16: row=lane-16, K = {8..15, 24..31}
__device__ __forceinline__ v16bf load_frag(const __bf16* __restrict__ p, int ld, int lane) {
  const int row = lane & 15;
  const int kb  = (lane >> 4) << 3;  // 0 or 8
  const __bf16* r0 = p + (size_t)row * ld + kb;
  v16bf f;
#pragma unroll
  for (int i = 0; i < 8; ++i) f[i] = r0[i];
#pragma unroll
  for (int i = 0; i < 8; ++i) f[8 + i] = r0[16 + i];
  return f;
}

// ---------------------------------------------------------------------------
// Generic GEMM: C[M,N] = A[M,K](bf16) * W[N,K](bf16)^T + bias[N]
// Block: 256 threads = 8 waves. Block tile 128(M) x 128(N).
// Wave w -> rows [w*16, w*16+16), all 128 cols (8 accumulators): each A
// fragment feeds 8 v_wmma. B tile (128x32 bf16 = 8KB) staged into LDS with
// GLOBAL_LOAD_ASYNC_TO_LDS_B128, shared by all 8 waves.
// ---------------------------------------------------------------------------
__global__ __launch_bounds__(256) void gemm_bf16_wmma(
    const __bf16* __restrict__ A, const __bf16* __restrict__ W,
    const float* __restrict__ bias, float* __restrict__ C,
    int M, int N, int K) {
  __shared__ __align__(16) __bf16 Btile[128 * 32];
  const int tid  = threadIdx.x;
  const int lane = tid & 31;
  const int wave = tid >> 5;
  const int m0 = blockIdx.x * 128 + wave * 16;
  const int n0 = blockIdx.y * 128;

  // staging: 128 rows x 4 chunks of 8 bf16 (16B) = 512 chunks, 2 per thread
  const int c0row = tid >> 2;                 // chunk tid   : row 0..63
  const int c0kk  = (tid & 3) * 8;
  const int c1row = (tid + 256) >> 2;         // chunk tid+256: row 64..127
  const int c1kk  = c0kk;
  const unsigned ldsBase = (unsigned)(uintptr_t)&Btile[0];

  v8f acc[8] = {};
  for (int k0 = 0; k0 < K; k0 += 32) {
    async_copy_b128(ldsBase + (unsigned)(c0row * 32 + c0kk) * 2,
                    W + (size_t)(n0 + c0row) * K + k0 + c0kk);
    async_copy_b128(ldsBase + (unsigned)(c1row * 32 + c1kk) * 2,
                    W + (size_t)(n0 + c1row) * K + k0 + c1kk);
    wait_async0();
    __syncthreads();

    v16bf af = load_frag(A + (size_t)m0 * K + k0, K, lane);
#pragma unroll
    for (int nt = 0; nt < 8; ++nt) {
      v16bf bfrag = load_frag(&Btile[nt * 16 * 32], 32, lane);
      acc[nt] = WMMA_BF16(af, bfrag, acc[nt]);
    }
    __syncthreads();
  }

  // C/D layout: VGPR r, lanes 0-15 -> M=r, lanes 16-31 -> M=8+r; N=lane&15
  const int rbase = m0 + ((lane >> 4) << 3);
#pragma unroll
  for (int nt = 0; nt < 8; ++nt) {
    const int col = n0 + nt * 16 + (lane & 15);
    const float bv = bias ? bias[col] : 0.0f;
#pragma unroll
    for (int r = 0; r < 8; ++r)
      C[(size_t)(rbase + r) * N + col] = acc[nt][r] + bv;
  }
}

// ---------------------------------------------------------------------------
// Fused LSTM timestep: G = hprev(64x512,bf16) * Whh(2048x512,bf16)^T via WMMA,
// then gates -> c,h. Grid = 8 blocks (n-chunks of 64), 256 thr = 8 waves.
// Each block computes G[64 x 256] (4 gates x 64 cols) in LDS.
// ---------------------------------------------------------------------------
__global__ __launch_bounds__(256) void lstm_step(
    const __bf16* __restrict__ hprev,   // [64,512]
    const __bf16* __restrict__ whh,     // [2048,512]
    const float*  __restrict__ xg,      // [B*T, 2048]  (x@Wih^T + b_ih)
    const float*  __restrict__ bhh,     // [2048]
    float*        __restrict__ cstate,  // [64,512]
    __bf16*       __restrict__ hnext,   // [64,512]
    float*        __restrict__ dec,     // [B*T, 512]
    __bf16*       __restrict__ cat,     // [B*T, 1024] (write cols 0..511)
    int t) {
  __shared__ float G[64 * 256];
  const int tid  = threadIdx.x;
  const int lane = tid & 31;
  const int wave = tid >> 5;
  const int n0 = blockIdx.x * 64;
  const int mtile  = wave >> 1;        // 0..3  (rows 16*mtile..+16)
  const int ntbase = (wave & 1) * 8;   // 8 n-tiles per wave

  v8f acc[8] = {};
  for (int k0 = 0; k0 < 512; k0 += 32) {
    v16bf af = load_frag(hprev + mtile * 16 * 512 + k0, 512, lane);
#pragma unroll
    for (int i = 0; i < 8; ++i) {
      const int ntil = ntbase + i;                                // 0..15
      const int wrow = (ntil >> 2) * 512 + n0 + (ntil & 3) * 16;  // gate*H + n
      v16bf bfrag = load_frag(whh + (size_t)wrow * 512 + k0, 512, lane);
      acc[i] = WMMA_BF16(af, bfrag, acc[i]);
    }
  }
  // spill gate pre-activations to LDS: col = gate*64 + local_n
  const int rb = mtile * 16 + ((lane >> 4) << 3);
#pragma unroll
  for (int i = 0; i < 8; ++i) {
    const int ntil = ntbase + i;
    const int col = (ntil >> 2) * 64 + (ntil & 3) * 16 + (lane & 15);
#pragma unroll
    for (int r = 0; r < 8; ++r) G[(rb + r) * 256 + col] = acc[i][r];
  }
  __syncthreads();

  // pointwise gates: 64 batches x 64 cols
  for (int it = 0; it < 16; ++it) {
    const int idx = tid + it * 256;
    const int b  = idx >> 6;
    const int nn = idx & 63;
    const int n  = n0 + nn;
    const size_t row = (size_t)b * Tt + t;
    const float gi = G[b * 256 + nn]       + xg[row * 2048 + n]        + bhh[n];
    const float gf = G[b * 256 + 64 + nn]  + xg[row * 2048 + 512 + n]  + bhh[512 + n];
    const float gg = G[b * 256 + 128 + nn] + xg[row * 2048 + 1024 + n] + bhh[1024 + n];
    const float go = G[b * 256 + 192 + nn] + xg[row * 2048 + 1536 + n] + bhh[1536 + n];
    const float c = sigmoidf_(gf) * cstate[b * 512 + n] + sigmoidf_(gi) * tanhf(gg);
    const float h = sigmoidf_(go) * tanhf(c);
    cstate[b * 512 + n] = c;
    hnext[b * 512 + n]  = (__bf16)h;
    dec[row * 512 + n]  = h;
    cat[row * 1024 + n] = (__bf16)h;  // left half of [dec || attn]
  }
}

// ---------------------------------------------------------------------------
// Fused attention per batch: scores = dec*enc^T + mask, softmax, ctx = p*enc.
// One block per b; scores live in LDS. Writes bf16 ctx into cat cols 512..1023.
// ---------------------------------------------------------------------------
__global__ __launch_bounds__(256) void attention(
    const float* __restrict__ dec,      // [B*T, 512]
    const float* __restrict__ enc,      // [B, S, 512]
    const int*   __restrict__ de_lens,  // [B]
    __bf16*      __restrict__ cat) {    // [B*T, 1024]
  __shared__ float sc[Tt * 61];
  const int b   = blockIdx.x;
  const int tid = threadIdx.x;
  const int len = de_lens[b];

  // scores
  for (int idx = tid; idx < Tt * Ss; idx += 256) {
    const int t = idx / Ss, s = idx % Ss;
    const float* dr = dec + ((size_t)b * Tt + t) * Hh;
    const float* er = enc + ((size_t)b * Ss + s) * Hh;
    float acc = 0.0f;
    for (int h = 0; h < Hh; ++h) acc += dr[h] * er[h];
    sc[t * 61 + s] = acc + (s < len ? 0.0f : NEGMASK);
  }
  __syncthreads();

  // softmax, one row per thread (t < 60)
  if (tid < Tt) {
    float* r = &sc[tid * 61];
    float mx = r[0];
    for (int s = 1; s < Ss; ++s) mx = fmaxf(mx, r[s]);
    float sum = 0.0f;
    for (int s = 0; s < Ss; ++s) { float e = __expf(r[s] - mx); r[s] = e; sum += e; }
    const float inv = 1.0f / sum;
    for (int s = 0; s < Ss; ++s) r[s] *= inv;
  }
  __syncthreads();

  // context
  for (int idx = tid; idx < Tt * Hh; idx += 256) {
    const int t = idx >> 9, h = idx & 511;
    float acc = 0.0f;
    const float* p = &sc[t * 61];
    const float* ec = enc + (size_t)b * Ss * Hh + h;
    for (int s = 0; s < Ss; ++s) acc += p[s] * ec[(size_t)s * Hh];
    cat[((size_t)b * Tt + t) * 1024 + 512 + h] = (__bf16)acc;
  }
}

// ---------------------------------------------------------------------------
// Small helpers
// ---------------------------------------------------------------------------
__global__ void f32_to_bf16(const float* __restrict__ in, __bf16* __restrict__ out, size_t n) {
  size_t i = (size_t)blockIdx.x * 256 + threadIdx.x;
  if (i < n) out[i] = (__bf16)in[i];
}

__global__ void embed_gather(const int* __restrict__ idx, const float* __restrict__ emb,
                             __bf16* __restrict__ xbf, int total) {
  int i = blockIdx.x * 256 + threadIdx.x;
  if (i < total) {
    const int row = i >> 9, h = i & 511;
    xbf[i] = (__bf16)emb[(size_t)idx[row] * Hh + h];
  }
}

__global__ void init_state(__bf16* __restrict__ h, float* __restrict__ c, int n) {
  int i = blockIdx.x * 256 + threadIdx.x;
  if (i < n) { h[i] = (__bf16)0.0f; c[i] = 0.0f; }
}

// ---------------------------------------------------------------------------
extern "C" void kernel_launch(void* const* d_in, const int* in_sizes, int n_in,
                              void* d_out, int out_size, void* d_ws, size_t ws_size,
                              hipStream_t stream) {
  (void)in_sizes; (void)n_in; (void)out_size; (void)ws_size;
  const int*   indices = (const int*)d_in[0];
  const float* enc     = (const float*)d_in[1];
  const int*   de_lens = (const int*)d_in[2];
  const float* emb     = (const float*)d_in[3];
  const float* w_ih    = (const float*)d_in[4];
  const float* w_hh    = (const float*)d_in[5];
  const float* b_ih    = (const float*)d_in[6];
  const float* b_hh    = (const float*)d_in[7];
  const float* lin_w   = (const float*)d_in[8];
  const float* lin_b   = (const float*)d_in[9];
  const float* out_w   = (const float*)d_in[10];
  const float* out_b   = (const float*)d_in[11];
  float* logits = (float*)d_out;

  char* wp = (char*)d_ws;
  auto take = [&](size_t bytes) { char* r = wp; wp += (bytes + 255) & ~(size_t)255; return r; };
  __bf16* wih_bf  = (__bf16*)take((size_t)Ll * 4 * Hh * Hh * 2);
  __bf16* whh_bf  = (__bf16*)take((size_t)Ll * 4 * Hh * Hh * 2);
  __bf16* linw_bf = (__bf16*)take((size_t)Ll * Hh * 2 * Hh * 2);
  __bf16* outw_bf = (__bf16*)take((size_t)Vv * Hh * 2);
  __bf16* xbf     = (__bf16*)take((size_t)Bb * Tt * Hh * 2);
  float*  xf32    = (float*)take((size_t)Bb * Tt * Hh * 4);
  float*  xg      = (float*)take((size_t)Bb * Tt * 4 * Hh * 4);
  float*  dec     = (float*)take((size_t)Bb * Tt * Hh * 4);
  __bf16* cat     = (__bf16*)take((size_t)Bb * Tt * 2 * Hh * 2);
  __bf16* hA      = (__bf16*)take((size_t)Bb * Hh * 2);
  __bf16* hB      = (__bf16*)take((size_t)Bb * Hh * 2);
  float*  cbuf    = (float*)take((size_t)Bb * Hh * 4);

  auto cvt = [&](const float* src, __bf16* dst, size_t n) {
    f32_to_bf16<<<dim3((unsigned)((n + 255) / 256)), dim3(256), 0, stream>>>(src, dst, n);
  };
  auto gemm = [&](const __bf16* A, const __bf16* W, const float* bias, float* C,
                  int M, int N, int K) {
    gemm_bf16_wmma<<<dim3(M / 128, N / 128), dim3(256), 0, stream>>>(A, W, bias, C, M, N, K);
  };

  // weights -> bf16 (deterministic each call)
  cvt(w_ih,  wih_bf,  (size_t)Ll * 4 * Hh * Hh);
  cvt(w_hh,  whh_bf,  (size_t)Ll * 4 * Hh * Hh);
  cvt(lin_w, linw_bf, (size_t)Ll * Hh * 2 * Hh);
  cvt(out_w, outw_bf, (size_t)Vv * Hh);

  // x = emb[indices] (bf16)
  {
    const int total = Bb * Tt * Hh;
    embed_gather<<<dim3((total + 255) / 256), dim3(256), 0, stream>>>(indices, emb, xbf, total);
  }

  const int M = Bb * Tt;  // 3840
  for (int l = 0; l < Ll; ++l) {
    const __bf16* wih_l = wih_bf + (size_t)l * 4 * Hh * Hh;
    const __bf16* whh_l = whh_bf + (size_t)l * 4 * Hh * Hh;
    const __bf16* lnw_l = linw_bf + (size_t)l * Hh * 2 * Hh;

    // xg = x @ Wih^T + b_ih   [3840, 2048]
    gemm(xbf, wih_l, b_ih + (size_t)l * 4 * Hh, xg, M, 4 * Hh, Hh);

    // LSTM recurrence
    init_state<<<dim3((Bb * Hh + 255) / 256), dim3(256), 0, stream>>>(hA, cbuf, Bb * Hh);
    __bf16 *hp = hA, *hn = hB;
    for (int t = 0; t < Tt; ++t) {
      lstm_step<<<dim3(8), dim3(256), 0, stream>>>(
          hp, whh_l, xg, b_hh + (size_t)l * 4 * Hh, cbuf, hn, dec, cat, t);
      __bf16* tmp = hp; hp = hn; hn = tmp;
    }

    // attention -> cat[:, 512:]
    attention<<<dim3(Bb), dim3(256), 0, stream>>>(dec, enc, de_lens, cat);

    // x = cat @ lin_w^T + lin_b   [3840, 512]
    gemm(cat, lnw_l, lin_b + (size_t)l * Hh, xf32, M, Hh, 2 * Hh);
    cvt(xf32, xbf, (size_t)M * Hh);
  }

  // logits = x @ out_w^T + out_b   [3840, 32000]
  gemm(xbf, outw_bf, out_b, logits, M, Vv, Hh);
}
